// AttnEncoderDecoder_7198365188431
// MI455X (gfx1250) — matrix operand order repaired and checked by reading, not compiled, not run on hardware
//
#include <hip/hip_runtime.h>
#include <math.h>

// ---------------------------------------------------------------------------
// CDNA5 (gfx1250) wave32 WMMA implementation of a pointer-generator seq2seq.
// Matrix work runs on v_wmma_f32_16x16x32_f16 (f16 in, f32 accumulate).
// ---------------------------------------------------------------------------

typedef __attribute__((ext_vector_type(16))) _Float16 v16h;
typedef __attribute__((ext_vector_type(8)))  _Float16 v8h;
typedef __attribute__((ext_vector_type(8)))  float    v8f;

#define BSZ   16
#define EMBD  128
#define HID   512
#define G3    1536          // 3*HID gate width
#define SEQ   400
#define TDEC  64
#define VOC   32000
#define NOOV  50
#define VEXT  (VOC + NOOV)  // 32050
#define TB    (TDEC * BSZ)  // 1024 decoder rows
#define START_TOK 2

// ---- WMMA tile loaders (16x16x32 f16, wave32; contiguous k order, same in A and B) --
// A (16x32, MxK): lane = half*16 + m. Elements e0..15 hold K = 16*half + (0..15)
// -> two 16B loads per lane.
__device__ __forceinline__ v16h load_a_tile(const _Float16* A, int lda, int lane) {
  const int m = lane & 15, hh = lane >> 4;
  union { v16h v; v8h h[2]; } u;
  const _Float16* p = A + (size_t)m * lda + hh * 16;
  u.h[0] = *(const v8h*)(p);
  u.h[1] = *(const v8h*)(p + 8);
  return u.v;
}

// B (32x16, KxN) built from torch-layout W[N,K] row-major (B = W^T):
// lanes 0-15 hold K=0..15, lanes 16-31 hold K=16..31, N = lane&15.
// Per lane: 16 contiguous halfs along K -> one 32B load.
__device__ __forceinline__ v16h load_b_tile(const _Float16* W, int ldw, int lane) {
  const int n = lane & 15, hh = lane >> 4;
  return *(const v16h*)(W + (size_t)n * ldw + hh * 16);
}

__device__ __forceinline__ v8f wmma_f16(v16h a, v16h b, v8f acc) {
  return __builtin_amdgcn_wmma_f32_16x16x32_f16(
      /*neg_a=*/false, a, /*neg_b=*/false, b,
      /*c_mod=*/(short)0, acc, /*reuse_a=*/false, /*reuse_b=*/false);
}

__device__ __forceinline__ float sigmoidf_(float x) { return 1.0f / (1.0f + expf(-x)); }

// ---------------------------------------------------------------------------
// f32 -> f16 weight conversion
// ---------------------------------------------------------------------------
__global__ void cvt_f32_f16_kernel(const float* __restrict__ s,
                                   _Float16* __restrict__ d, int n) {
  int i = blockIdx.x * 256 + threadIdx.x;
  if (i < n) d[i] = (_Float16)s[i];
}

// ---------------------------------------------------------------------------
// GRU step core, shared by encoder/decoder:
// 32 waves, wave w owns hidden columns [w*16, w*16+16) for ALL three gates.
// r,z accumulate gi+gh into a single WMMA chain each (bias pre-loaded into C);
// n keeps inn / hn separate.  4 accumulators, no global gates round-trip.
// ---------------------------------------------------------------------------
struct GruBias { float br, bz, bni, bnh; };

__device__ __forceinline__ GruBias gru_load_bias(const float* bih, const float* bhh,
                                                 int c) {
  GruBias g;
  g.br  = bih[c]           + bhh[c];
  g.bz  = bih[HID + c]     + bhh[HID + c];
  g.bni = bih[2 * HID + c];
  g.bnh = bhh[2 * HID + c];
  return g;
}

// ---------------------------------------------------------------------------
// Bidirectional encoder GRU: grid = 2 blocks (dir 0 fwd, dir 1 bwd),
// 1024 threads = 32 waves, 400 sequential steps.
// ---------------------------------------------------------------------------
__global__ __launch_bounds__(1024) void encoder_gru_kernel(
    const int* __restrict__ batch_X, const float* __restrict__ E_enc,
    const _Float16* __restrict__ WihF, const _Float16* __restrict__ WhhF,
    const float* __restrict__ bihF, const float* __restrict__ bhhF,
    const _Float16* __restrict__ WihB, const _Float16* __restrict__ WhhB,
    const float* __restrict__ bihB, const float* __restrict__ bhhB,
    _Float16* __restrict__ enc_cat,   // [SEQ][BSZ][1024]  (fwd 0:512, bwd 512:1024)
    _Float16* __restrict__ hcat)      // [BSZ][1024]       final [hf, hb]
{
  const int dir = blockIdx.x;
  const _Float16* Wih = dir ? WihB : WihF;
  const _Float16* Whh = dir ? WhhB : WhhF;
  const float* bih = dir ? bihB : bihF;
  const float* bhh = dir ? bhhB : bhhF;

  __shared__ _Float16 x16[BSZ][EMBD];
  __shared__ _Float16 h16[BSZ][HID];
  __shared__ float    h32[BSZ][HID];

  const int tid = threadIdx.x, lane = tid & 31, wid = tid >> 5;
  const int n = lane & 15, mh = (lane >> 4) << 3;
  const int c0 = wid * 16;               // this wave's hidden-column tile
  const GruBias gb = gru_load_bias(bih, bhh, c0 + n);

  for (int i = tid; i < BSZ * HID; i += 1024) {
    (&h32[0][0])[i] = 0.0f;
    (&h16[0][0])[i] = (_Float16)0.0f;
  }
  __syncthreads();

  for (int step = 0; step < SEQ; ++step) {
    const int t = dir ? (SEQ - 1 - step) : step;

    // embedding gather -> f16 LDS
    for (int i = tid; i < BSZ * EMBD; i += 1024) {
      int b = i >> 7, e = i & (EMBD - 1);
      int tok = batch_X[t * BSZ + b];
      x16[b][e] = (_Float16)E_enc[(size_t)tok * EMBD + e];
    }
    __syncthreads();
    // keep weight loads inside the step loop (stream from L2, never spill)
    asm volatile("" ::: "memory");

    v8f accr, accz, accni, accnh;
    #pragma unroll
    for (int j = 0; j < 8; ++j) {
      accr[j] = gb.br; accz[j] = gb.bz; accni[j] = gb.bni; accnh[j] = gb.bnh;
    }
    #pragma unroll
    for (int k0 = 0; k0 < EMBD; k0 += 32) {          // x @ Wih^T (r,z,n)
      v16h a = load_a_tile(&x16[0][0] + k0, EMBD, lane);
      accr  = wmma_f16(a, load_b_tile(Wih + (size_t)(c0)           * EMBD + k0, EMBD, lane), accr);
      accz  = wmma_f16(a, load_b_tile(Wih + (size_t)(HID + c0)     * EMBD + k0, EMBD, lane), accz);
      accni = wmma_f16(a, load_b_tile(Wih + (size_t)(2 * HID + c0) * EMBD + k0, EMBD, lane), accni);
    }
    #pragma unroll
    for (int k0 = 0; k0 < HID; k0 += 32) {           // h @ Whh^T (r,z,n)
      v16h a = load_a_tile(&h16[0][0] + k0, HID, lane);
      accr  = wmma_f16(a, load_b_tile(Whh + (size_t)(c0)           * HID + k0, HID, lane), accr);
      accz  = wmma_f16(a, load_b_tile(Whh + (size_t)(HID + c0)     * HID + k0, HID, lane), accz);
      accnh = wmma_f16(a, load_b_tile(Whh + (size_t)(2 * HID + c0) * HID + k0, HID, lane), accnh);
    }
    __syncthreads();   // all waves done reading h16/x16

    // GRU update entirely in registers (C/D layout: row mh+j, col c0+n)
    #pragma unroll
    for (int j = 0; j < 8; ++j) {
      const int bb = mh + j, cc = c0 + n;
      float r  = sigmoidf_(accr[j]);
      float z  = sigmoidf_(accz[j]);
      float nn = tanhf(accni[j] + r * accnh[j]);
      float h2 = (1.0f - z) * nn + z * h32[bb][cc];
      h32[bb][cc] = h2;
      h16[bb][cc] = (_Float16)h2;
      enc_cat[((size_t)t * BSZ + bb) * 1024 + (size_t)dir * HID + cc] = (_Float16)h2;
    }
    __syncthreads();   // h ready for next step
  }

  // final hidden -> [hf, hb]
  for (int i = tid; i < BSZ * HID; i += 1024) {
    int b = i >> 9, c = i & (HID - 1);
    hcat[b * 1024 + dir * HID + c] = (_Float16)h32[b][c];
  }
}

// ---------------------------------------------------------------------------
// Decoder GRU: single block, T=64 sequential steps (teacher forcing).
// Writes states f32 (for p_gen) and f16 into left half of concat buffer.
// ---------------------------------------------------------------------------
__global__ __launch_bounds__(1024) void decoder_gru_kernel(
    const int* __restrict__ batch_Y, const float* __restrict__ E_dec,
    const _Float16* __restrict__ Wih, const _Float16* __restrict__ Whh,
    const float* __restrict__ bih, const float* __restrict__ bhh,
    const float* __restrict__ h0,      // [BSZ][HID] relu(bi2uni(...))
    float* __restrict__ dec_h,         // [TB][HID]
    _Float16* __restrict__ concat_in)  // [TB][1024], left half = h2
{
  __shared__ _Float16 x16[BSZ][EMBD];
  __shared__ _Float16 h16[BSZ][HID];
  __shared__ float    h32[BSZ][HID];

  const int tid = threadIdx.x, lane = tid & 31, wid = tid >> 5;
  const int n = lane & 15, mh = (lane >> 4) << 3;
  const int c0 = wid * 16;
  const GruBias gb = gru_load_bias(bih, bhh, c0 + n);

  for (int i = tid; i < BSZ * HID; i += 1024) {
    float v = h0[i];
    (&h32[0][0])[i] = v;
    (&h16[0][0])[i] = (_Float16)v;
  }
  __syncthreads();

  for (int t = 0; t < TDEC; ++t) {
    for (int i = tid; i < BSZ * EMBD; i += 1024) {
      int b = i >> 7, e = i & (EMBD - 1);
      int tok = (t == 0) ? START_TOK : batch_Y[(t - 1) * BSZ + b];
      x16[b][e] = (_Float16)E_dec[(size_t)tok * EMBD + e];
    }
    __syncthreads();
    asm volatile("" ::: "memory");

    v8f accr, accz, accni, accnh;
    #pragma unroll
    for (int j = 0; j < 8; ++j) {
      accr[j] = gb.br; accz[j] = gb.bz; accni[j] = gb.bni; accnh[j] = gb.bnh;
    }
    #pragma unroll
    for (int k0 = 0; k0 < EMBD; k0 += 32) {
      v16h a = load_a_tile(&x16[0][0] + k0, EMBD, lane);
      accr  = wmma_f16(a, load_b_tile(Wih + (size_t)(c0)           * EMBD + k0, EMBD, lane), accr);
      accz  = wmma_f16(a, load_b_tile(Wih + (size_t)(HID + c0)     * EMBD + k0, EMBD, lane), accz);
      accni = wmma_f16(a, load_b_tile(Wih + (size_t)(2 * HID + c0) * EMBD + k0, EMBD, lane), accni);
    }
    #pragma unroll
    for (int k0 = 0; k0 < HID; k0 += 32) {
      v16h a = load_a_tile(&h16[0][0] + k0, HID, lane);
      accr  = wmma_f16(a, load_b_tile(Whh + (size_t)(c0)           * HID + k0, HID, lane), accr);
      accz  = wmma_f16(a, load_b_tile(Whh + (size_t)(HID + c0)     * HID + k0, HID, lane), accz);
      accnh = wmma_f16(a, load_b_tile(Whh + (size_t)(2 * HID + c0) * HID + k0, HID, lane), accnh);
    }
    __syncthreads();

    #pragma unroll
    for (int j = 0; j < 8; ++j) {
      const int bb = mh + j, cc = c0 + n;
      float r  = sigmoidf_(accr[j]);
      float z  = sigmoidf_(accz[j]);
      float nn = tanhf(accni[j] + r * accnh[j]);
      float h2 = (1.0f - z) * nn + z * h32[bb][cc];
      h32[bb][cc] = h2;
      h16[bb][cc] = (_Float16)h2;
      size_t tb = (size_t)t * BSZ + bb;
      dec_h[tb * HID + cc] = h2;
      concat_in[tb * 1024 + cc] = (_Float16)h2;
    }
    __syncthreads();
  }
}

// ---------------------------------------------------------------------------
// Generic WMMA GEMM: C[M,N] = act(A[M,K] @ W[N,K]^T + bias)
// 8 waves/block; each wave computes a 16x64 strip (4 accumulators) so every
// A tile feeds 4 WMMAs.  act: 0=none, 1=relu, 2=tanh.  f32 and/or f16 out.
// ---------------------------------------------------------------------------
__global__ __launch_bounds__(256) void wmma_gemm_kernel(
    const _Float16* __restrict__ A, int lda,
    const _Float16* __restrict__ W, int ldw,
    const float* __restrict__ bias, int act,
    float* __restrict__ Cf, int ldc,
    _Float16* __restrict__ Ch, int ldch,
    int M, int N, int K)
{
  const int wid = threadIdx.x >> 5, lane = threadIdx.x & 31;
  const int nstrips = N >> 6;               // strips of 64 columns
  const long job   = (long)blockIdx.x * 8 + wid;
  const long total = (long)(M >> 4) * nstrips;
  if (job >= total) return;                 // wave-uniform -> EXEC stays all-ones
  const int mt = (int)(job / nstrips);
  const int ns = (int)(job % nstrips);

  const _Float16* Ap = A + (size_t)mt * 16 * lda;
  const _Float16* Wp = W + (size_t)ns * 64 * ldw;

  v8f acc[4] = {{}, {}, {}, {}};
  for (int k0 = 0; k0 < K; k0 += 32) {
    if (k0 + 32 < K)  // warm caches for the A stream (global_prefetch_b8)
      __builtin_prefetch(Ap + (size_t)(lane & 15) * lda + k0 + 32, 0, 1);
    v16h a = load_a_tile(Ap + k0, lda, lane);
    #pragma unroll
    for (int j = 0; j < 4; ++j)
      acc[j] = wmma_f16(a, load_b_tile(Wp + (size_t)j * 16 * ldw + k0, ldw, lane), acc[j]);
  }

  const int n = lane & 15, mh = (lane >> 4) << 3;
  #pragma unroll
  for (int j = 0; j < 4; ++j) {
    const size_t col = (size_t)ns * 64 + j * 16 + n;
    const float bv = bias ? bias[col] : 0.0f;
    #pragma unroll
    for (int r = 0; r < 8; ++r) {
      float v = acc[j][r] + bv;
      if (act == 1)      v = fmaxf(v, 0.0f);
      else if (act == 2) v = tanhf(v);
      const size_t row = (size_t)mt * 16 + mh + r;
      if (Cf) Cf[row * (size_t)ldc  + col] = v;
      if (Ch) Ch[row * (size_t)ldch + col] = (_Float16)v;
    }
  }
}

// ---------------------------------------------------------------------------
// Attention per (t,b): energies tanh(hq+enc_proj)·v, softmax over S,
// context, p_gen gate. One 256-thread block per (t,b).
// ---------------------------------------------------------------------------
__global__ __launch_bounds__(256) void attn_kernel(
    const float* __restrict__ hq,        // [TB][HID] (includes attn_h_b)
    const float* __restrict__ enc_proj,  // [SEQ][BSZ][HID] (includes attn_e_b)
    const float* __restrict__ enc_out,   // [SEQ][BSZ][HID]
    const float* __restrict__ attn_v_W, const float* __restrict__ attn_v_b,
    const float* __restrict__ dec_h,     // [TB][HID]
    const int* __restrict__ batch_Y, const float* __restrict__ E_dec,
    const float* __restrict__ pg_c_W, const float* __restrict__ pg_c_b,
    const float* __restrict__ pg_e_W, const float* __restrict__ pg_e_b,
    const float* __restrict__ pg_r_W, const float* __restrict__ pg_r_b,
    _Float16* __restrict__ concat_in,    // [TB][1024], right half = ctx
    float* __restrict__ a_out,           // [TB][SEQ]
    float* __restrict__ pgen_out)        // [TB]
{
  const int tb = blockIdx.x;
  const int t = tb >> 4, b = tb & 15;
  const int tid = threadIdx.x;

  __shared__ float hq_s[HID], v_s[HID], ctx_s[HID], a_s[SEQ], red[256];

  for (int i = tid; i < HID; i += 256) {
    hq_s[i] = hq[(size_t)tb * HID + i];
    v_s[i]  = attn_v_W[i];
  }
  __syncthreads();

  for (int s = tid; s < SEQ; s += 256) {
    const float* ep = enc_proj + ((size_t)s * BSZ + b) * HID;
    float acc = 0.0f;
    for (int h = 0; h < HID; ++h) acc += tanhf(hq_s[h] + ep[h]) * v_s[h];
    a_s[s] = acc + attn_v_b[0];
  }
  __syncthreads();

  // softmax over SEQ
  float lmax = -1.0e30f;
  for (int s = tid; s < SEQ; s += 256) lmax = fmaxf(lmax, a_s[s]);
  red[tid] = lmax; __syncthreads();
  for (int off = 128; off > 0; off >>= 1) {
    if (tid < off) red[tid] = fmaxf(red[tid], red[tid + off]);
    __syncthreads();
  }
  const float mx = red[0]; __syncthreads();

  float lsum = 0.0f;
  for (int s = tid; s < SEQ; s += 256) {
    float e = expf(a_s[s] - mx);
    a_s[s] = e; lsum += e;
  }
  red[tid] = lsum; __syncthreads();
  for (int off = 128; off > 0; off >>= 1) {
    if (tid < off) red[tid] += red[tid + off];
    __syncthreads();
  }
  const float inv = 1.0f / red[0]; __syncthreads();
  for (int s = tid; s < SEQ; s += 256) {
    a_s[s] *= inv;
    a_out[(size_t)tb * SEQ + s] = a_s[s];
  }
  __syncthreads();

  // context = sum_s a[s] * enc_out[s,b,:]
  for (int h = tid; h < HID; h += 256) {
    float acc = 0.0f;
    for (int s = 0; s < SEQ; ++s)
      acc += a_s[s] * enc_out[((size_t)s * BSZ + b) * HID + h];
    ctx_s[h] = acc;
    concat_in[(size_t)tb * 1024 + HID + h] = (_Float16)acc;
  }
  __syncthreads();

  // p_gen = sigmoid(ctx·pg_c + e·pg_e + h2·pg_r + biases)
  const int word = (t == 0) ? START_TOK : batch_Y[(t - 1) * BSZ + b];
  float acc = 0.0f;
  for (int h = tid; h < HID; h += 256)
    acc += ctx_s[h] * pg_c_W[h] + dec_h[(size_t)tb * HID + h] * pg_r_W[h];
  for (int e = tid; e < EMBD; e += 256)
    acc += E_dec[(size_t)word * EMBD + e] * pg_e_W[e];
  red[tid] = acc; __syncthreads();
  for (int off = 128; off > 0; off >>= 1) {
    if (tid < off) red[tid] += red[tid + off];
    __syncthreads();
  }
  if (tid == 0)
    pgen_out[tb] = sigmoidf_(red[0] + pg_c_b[0] + pg_e_b[0] + pg_r_b[0]);
}

// ---------------------------------------------------------------------------
// Finalize per (t,b) row of d_out: softmax(logits)*p_gen, zero OOV region,
// scatter-add (1-p_gen)*a onto articles_extend columns, then log(x + eps).
// ---------------------------------------------------------------------------
__global__ __launch_bounds__(256) void finalize_kernel(
    float* __restrict__ out,             // [TB][VEXT], logits in cols [0,VOC)
    const float* __restrict__ pgen,      // [TB]
    const float* __restrict__ a_out,     // [TB][SEQ]
    const int* __restrict__ articles)    // [BSZ][SEQ]
{
  const int tb = blockIdx.x;
  const int b = tb & 15;
  const int tid = threadIdx.x;
  float* row = out + (size_t)tb * VEXT;

  __shared__ float red[256];
  const float p = pgen[tb];

  float lmax = -1.0e30f;
  for (int i = tid; i < VOC; i += 256) lmax = fmaxf(lmax, row[i]);
  red[tid] = lmax; __syncthreads();
  for (int off = 128; off > 0; off >>= 1) {
    if (tid < off) red[tid] = fmaxf(red[tid], red[tid + off]);
    __syncthreads();
  }
  const float mx = red[0]; __syncthreads();

  float lsum = 0.0f;
  for (int i = tid; i < VOC; i += 256) {
    float e = expf(row[i] - mx);
    row[i] = e; lsum += e;
  }
  red[tid] = lsum; __syncthreads();
  for (int off = 128; off > 0; off >>= 1) {
    if (tid < off) red[tid] += red[tid + off];
    __syncthreads();
  }
  const float scale = p / red[0];

  for (int i = tid; i < VOC; i += 256) row[i] *= scale;
  for (int i = VOC + tid; i < VEXT; i += 256) row[i] = 0.0f;
  __threadfence();
  __syncthreads();

  // scatter-add copy distribution (duplicate columns accumulate)
  const float q = 1.0f - p;
  for (int s = tid; s < SEQ; s += 256) {
    int col = articles[b * SEQ + s];
    atomicAdd(&row[col], q * a_out[(size_t)tb * SEQ + s]);
  }
  __threadfence();
  __syncthreads();

  for (int i = tid; i < VEXT; i += 256) row[i] = logf(row[i] + 1e-12f);
}

// ---------------------------------------------------------------------------
// Host launcher
// ---------------------------------------------------------------------------
extern "C" void kernel_launch(void* const* d_in, const int* in_sizes, int n_in,
                              void* d_out, int out_size, void* d_ws, size_t ws_size,
                              hipStream_t stream) {
  (void)in_sizes; (void)n_in; (void)out_size; (void)ws_size;

  // setup_inputs() dict order, params flattened in insertion order
  const int*   batch_X   = (const int*)d_in[0];
  const int*   batch_Y   = (const int*)d_in[1];
  const int*   articles  = (const int*)d_in[2];
  // d_in[3]=max_length, d_in[4]=max_oovs (compile-time constants here)
  const float* E_enc     = (const float*)d_in[5];
  const float* enc_f_Wih = (const float*)d_in[6];
  const float* enc_f_Whh = (const float*)d_in[7];
  const float* enc_f_bih = (const float*)d_in[8];
  const float* enc_f_bhh = (const float*)d_in[9];
  const float* enc_b_Wih = (const float*)d_in[10];
  const float* enc_b_Whh = (const float*)d_in[11];
  const float* enc_b_bih = (const float*)d_in[12];
  const float* enc_b_bhh = (const float*)d_in[13];
  const float* enc_lin_W = (const float*)d_in[14];
  const float* enc_lin_b = (const float*)d_in[15];
  const float* bi2uni_W  = (const float*)d_in[16];
  const float* bi2uni_b  = (const float*)d_in[17];
  const float* E_dec     = (const float*)d_in[18];
  const float* dec_Wih   = (const float*)d_in[19];
  const float* dec_Whh   = (const float*)d_in[20];
  const float* dec_bih   = (const float*)d_in[21];
  const float* dec_bhh   = (const float*)d_in[22];
  const float* attn_h_W  = (const float*)d_in[23];
  const float* attn_h_b  = (const float*)d_in[24];
  const float* attn_e_W  = (const float*)d_in[25];
  const float* attn_e_b  = (const float*)d_in[26];
  const float* attn_v_W  = (const float*)d_in[27];
  const float* attn_v_b  = (const float*)d_in[28];
  const float* lin_W     = (const float*)d_in[29];
  const float* lin_b     = (const float*)d_in[30];
  const float* out_W     = (const float*)d_in[31];
  const float* out_b     = (const float*)d_in[32];
  const float* pg_c_W    = (const float*)d_in[33];
  const float* pg_c_b    = (const float*)d_in[34];
  const float* pg_e_W    = (const float*)d_in[35];
  const float* pg_e_b    = (const float*)d_in[36];
  const float* pg_r_W    = (const float*)d_in[37];
  const float* pg_r_b    = (const float*)d_in[38];

  // ---- workspace bump allocator (256B aligned) ----
  char* base = (char*)d_ws;
  size_t off = 0;
  auto alloc = [&](size_t elems, size_t elsz) -> void* {
    void* p = base + off;
    off += (elems * elsz + 255) & ~(size_t)255;
    return p;
  };

  _Float16* wfi       = (_Float16*)alloc((size_t)G3 * EMBD, 2);
  _Float16* wfh       = (_Float16*)alloc((size_t)G3 * HID, 2);
  _Float16* wbi       = (_Float16*)alloc((size_t)G3 * EMBD, 2);
  _Float16* wbh       = (_Float16*)alloc((size_t)G3 * HID, 2);
  _Float16* wdi       = (_Float16*)alloc((size_t)G3 * EMBD, 2);
  _Float16* wdh       = (_Float16*)alloc((size_t)G3 * HID, 2);
  _Float16* w_enclin  = (_Float16*)alloc((size_t)HID * 1024, 2);
  _Float16* w_bi2uni  = (_Float16*)alloc((size_t)HID * 1024, 2);
  _Float16* w_attnh   = (_Float16*)alloc((size_t)HID * HID, 2);
  _Float16* w_attne   = (_Float16*)alloc((size_t)HID * HID, 2);
  _Float16* w_lin     = (_Float16*)alloc((size_t)HID * 1024, 2);
  _Float16* w_out     = (_Float16*)alloc((size_t)VOC * HID, 2);
  _Float16* enc_cat   = (_Float16*)alloc((size_t)SEQ * BSZ * 1024, 2);
  _Float16* enc_out16 = (_Float16*)alloc((size_t)SEQ * BSZ * HID, 2);
  _Float16* hcat      = (_Float16*)alloc((size_t)BSZ * 1024, 2);
  _Float16* concat_in = (_Float16*)alloc((size_t)TB * 1024, 2);
  _Float16* concat_th = (_Float16*)alloc((size_t)TB * HID, 2);
  float* enc_out32 = (float*)alloc((size_t)SEQ * BSZ * HID, 4);
  float* enc_proj  = (float*)alloc((size_t)SEQ * BSZ * HID, 4);
  float* hidden0   = (float*)alloc((size_t)BSZ * HID, 4);
  float* dec_h     = (float*)alloc((size_t)TB * HID, 4);
  float* hq        = (float*)alloc((size_t)TB * HID, 4);
  float* a_buf     = (float*)alloc((size_t)TB * SEQ, 4);
  float* pgen      = (float*)alloc((size_t)TB, 4);

  // ---- weight conversions f32 -> f16 ----
  struct Job { const float* s; _Float16* d; int n; } jobs[] = {
    { enc_f_Wih, wfi,      G3 * EMBD }, { enc_f_Whh, wfh,      G3 * HID },
    { enc_b_Wih, wbi,      G3 * EMBD }, { enc_b_Whh, wbh,      G3 * HID },
    { dec_Wih,   wdi,      G3 * EMBD }, { dec_Whh,   wdh,      G3 * HID },
    { enc_lin_W, w_enclin, HID * 1024 }, { bi2uni_W, w_bi2uni, HID * 1024 },
    { attn_h_W,  w_attnh,  HID * HID }, { attn_e_W,  w_attne,  HID * HID },
    { lin_W,     w_lin,    HID * 1024 }, { out_W,    w_out,    VOC * HID },
  };
  for (int j = 0; j < 12; ++j) {
    int blocks = (jobs[j].n + 255) / 256;
    cvt_f32_f16_kernel<<<dim3(blocks), dim3(256), 0, stream>>>(
        jobs[j].s, jobs[j].d, jobs[j].n);
  }

  // ---- encoder (fwd + bwd concurrently) ----
  encoder_gru_kernel<<<dim3(2), dim3(1024), 0, stream>>>(
      batch_X, E_enc, wfi, wfh, enc_f_bih, enc_f_bhh,
      wbi, wbh, enc_b_bih, enc_b_bhh, enc_cat, hcat);

  auto gemm = [&](const _Float16* A, int lda, const _Float16* W, int ldw,
                  const float* bias, int act, float* Cf, int ldc,
                  _Float16* Ch, int ldch, int M, int N, int K) {
    long jobsn = (long)(M / 16) * (N / 64);
    int blocks = (int)((jobsn + 7) / 8);
    wmma_gemm_kernel<<<dim3(blocks), dim3(256), 0, stream>>>(
        A, lda, W, ldw, bias, act, Cf, ldc, Ch, ldch, M, N, K);
  };

  // enc_out = enc_cat @ enc_lin_W^T + b          [6400 x 512], K=1024
  gemm(enc_cat, 1024, w_enclin, 1024, enc_lin_b, 0,
       enc_out32, HID, enc_out16, HID, SEQ * BSZ, HID, 1024);
  // enc_proj = enc_out @ attn_e_W^T + b          [6400 x 512], K=512
  gemm(enc_out16, HID, w_attne, HID, attn_e_b, 0,
       enc_proj, HID, (_Float16*)nullptr, 0, SEQ * BSZ, HID, HID);
  // hidden = relu([hf,hb] @ bi2uni^T + b)        [16 x 512], K=1024
  gemm(hcat, 1024, w_bi2uni, 1024, bi2uni_b, 1,
       hidden0, HID, (_Float16*)nullptr, 0, BSZ, HID, 1024);

  // ---- decoder recurrence (only the truly sequential part) ----
  decoder_gru_kernel<<<dim3(1), dim3(1024), 0, stream>>>(
      batch_Y, E_dec, wdi, wdh, dec_bih, dec_bhh,
      hidden0, dec_h, concat_in);

  // hq = h2 @ attn_h_W^T + b over all T*B rows   [1024 x 512], K=512
  gemm(concat_in, 1024, w_attnh, HID, attn_h_b, 0,
       hq, HID, (_Float16*)nullptr, 0, TB, HID, HID);

  // attention / context / p_gen, batched over (t,b)
  attn_kernel<<<dim3(TB), dim3(256), 0, stream>>>(
      hq, enc_proj, enc_out32, attn_v_W, attn_v_b, dec_h, batch_Y, E_dec,
      pg_c_W, pg_c_b, pg_e_W, pg_e_b, pg_r_W, pg_r_b,
      concat_in, a_buf, pgen);

  // concat_tanh = tanh([h2,ctx] @ lin_W^T + b)   [1024 x 512], K=1024
  gemm(concat_in, 1024, w_lin, 1024, lin_b, 2,
       (float*)nullptr, 0, concat_th, HID, TB, HID, 1024);

  // logits = concat_tanh @ out_W^T + out_b  -> d_out rows (ldc = 32050)
  gemm(concat_th, HID, w_out, HID, out_b, 0,
       (float*)d_out, VEXT, (_Float16*)nullptr, 0, TB, VOC, HID);

  // softmax * p_gen, zero OOV, pointer scatter-add, log
  finalize_kernel<<<dim3(TB), dim3(256), 0, stream>>>(
      (float*)d_out, pgen, a_buf, articles);
}
